// SpatialGraphConv_part_90537910599794
// MI455X (gfx1250) — matrix operand order repaired and checked by reading, not compiled, not run on hardware
//
#include <hip/hip_runtime.h>

// CDNA5 / gfx1250: wave32, WMMA f32_16x16x32_f16 path.
typedef _Float16 v16h __attribute__((ext_vector_type(16)));
typedef _Float16 v8h  __attribute__((ext_vector_type(8)));
typedef float    v8f  __attribute__((ext_vector_type(8)));
typedef float    v2f  __attribute__((ext_vector_type(2)));

#define N_N    64
#define C_IN   64
#define T_T    300
#define V_V    25
#define TV     7500          // T*V, column dimension per n
#define C_OUT  128
#define KKK    3
#define NCOLT  469           // ceil(7500/16)
#define PITCH  72            // halves per LDS column row (64 + 8 pad); 144B, 16B aligned

// ---------------- prep: s[k][v] and beff[c][v] ----------------
__global__ __launch_bounds__(256) void sgc_prep(
    const float* __restrict__ A, const float* __restrict__ bias,
    const int* __restrict__ index, float* __restrict__ S, float* __restrict__ BEFF)
{
    __shared__ float s_sh[KKK * V_V];
    const int tid = threadIdx.x;
    if (tid < KKK * V_V) {
        const int k = tid / V_V, v = tid % V_V;
        const int idx = index[v];
        const float* row = A + (size_t)(k * V_V + idx) * V_V;
        float sum = 0.f;
        #pragma unroll
        for (int w = 0; w < V_V; ++w) sum += row[w];
        s_sh[tid] = sum;
        S[tid] = sum;
    }
    __syncthreads();
    for (int e = tid; e < C_OUT * V_V; e += 256) {
        const int c = e / V_V, v = e % V_V;
        BEFF[e] = s_sh[v]         * bias[c]
                + s_sh[V_V + v]   * bias[C_OUT + c]
                + s_sh[2*V_V + v] * bias[2*C_OUT + c];
    }
}

// ---------------- main GEMM + k-reduction ----------------
// grid = (column-tile groups, n). 8 waves/block; wave w owns output rows
// [w*16, w*16+16) of the 128 c-channels, for all 3 k-slices.
__global__ __launch_bounds__(256) void sgc_main(
    const float* __restrict__ x, const float* __restrict__ W,
    const float* __restrict__ S, const float* __restrict__ BEFF,
    float* __restrict__ out)
{
    __shared__ _Float16 tile[16 * PITCH];   // [col 0..15][K 0..63], f16

    const int tid  = threadIdx.x;
    const int lane = tid & 31;
    const int wid  = tid >> 5;       // c-tile 0..7
    const int m    = lane & 15;
    const int hi   = lane >> 4;      // 0 or 1
    const int n    = blockIdx.y;

    // A-operand (weights) in registers, contiguous K layout matching B:
    // v16h element j -> K = 16*hi + j.
    v16h aop[KKK][2];
    #pragma unroll
    for (int k = 0; k < KKK; ++k) {
        const float* wrow = W + (size_t)(k * C_OUT + wid * 16 + m) * C_IN;
        #pragma unroll
        for (int s = 0; s < 2; ++s) {
            v16h a;
            #pragma unroll
            for (int jj = 0; jj < 8; ++jj) {
                const int j  = jj * 2;
                const int Kg = s * 32 + 16 * hi + j;
                const v2f w2 = *(const v2f*)(wrow + Kg);   // consecutive K pair
                a[j]     = (_Float16)w2.x;
                a[j + 1] = (_Float16)w2.y;
            }
            aop[k][s] = a;
        }
    }

    const float* xn  = x   + (size_t)n * C_IN  * TV;
    float*       outn = out + (size_t)n * C_OUT * TV;

    for (int ct = blockIdx.x; ct < NCOLT; ct += gridDim.x) {
        const int colbase = ct * 16;

        __syncthreads();                       // prior consumers done with LDS
        {   // produce: 64x16 f32 -> f16 tile[col][K]; coalesced 64B rows
            const int c  = tid & 15;
            const int i0 = tid >> 4;           // 0..15
            const int gc0 = colbase + c;
            const int gc  = (gc0 < TV) ? gc0 : (TV - 1);
            #pragma unroll
            for (int r = 0; r < 4; ++r) {
                const int i = i0 + r * 16;
                tile[c * PITCH + i] = (_Float16)xn[(size_t)i * TV + gc];
            }
        }
        __syncthreads();

        // consume: B operand, contiguous K as in A: lanes 0-15 hold K=0..15, lanes 16-31
        // hold K=16..31 of each 32-K step. 16B-aligned ds_load_b128 reads.
        const _Float16* tp = &tile[m * PITCH + 16 * hi];
        const v8h b0 = *(const v8h*)(tp + 0);    // K = 16*hi + 0..7
        const v8h b1 = *(const v8h*)(tp + 8);    // K = 16*hi + 8..15
        const v8h b2 = *(const v8h*)(tp + 32);   // K = 32 + 16*hi + 0..7
        const v8h b3 = *(const v8h*)(tp + 40);   // K = 32 + 16*hi + 8..15
        const v16h B0 = __builtin_shufflevector(b0, b1, 0,1,2,3,4,5,6,7,8,9,10,11,12,13,14,15);
        const v16h B1 = __builtin_shufflevector(b2, b3, 0,1,2,3,4,5,6,7,8,9,10,11,12,13,14,15);

        v8f acc0 = {}, acc1 = {}, acc2 = {};
        acc0 = __builtin_amdgcn_wmma_f32_16x16x32_f16(false, aop[0][0], false, B0, (short)0, acc0, false, false);
        acc0 = __builtin_amdgcn_wmma_f32_16x16x32_f16(false, aop[0][1], false, B1, (short)0, acc0, false, false);
        acc1 = __builtin_amdgcn_wmma_f32_16x16x32_f16(false, aop[1][0], false, B0, (short)0, acc1, false, false);
        acc1 = __builtin_amdgcn_wmma_f32_16x16x32_f16(false, aop[1][1], false, B1, (short)0, acc1, false, false);
        acc2 = __builtin_amdgcn_wmma_f32_16x16x32_f16(false, aop[2][0], false, B0, (short)0, acc2, false, false);
        acc2 = __builtin_amdgcn_wmma_f32_16x16x32_f16(false, aop[2][1], false, B1, (short)0, acc2, false, false);

        // epilogue: out = s0*acc0 + s1*acc1 + s2*acc2 + beff[c][v]
        const int col = colbase + m;
        const int v   = col % V_V;
        const float s0 = S[v], s1 = S[V_V + v], s2 = S[2 * V_V + v];
        const bool ok = (col < TV);
        #pragma unroll
        for (int r = 0; r < 8; ++r) {
            const int M = r + 8 * hi;            // D layout: lanes16-31 -> M+8
            const int c = wid * 16 + M;
            const float val = s0 * acc0[r] + s1 * acc1[r] + s2 * acc2[r]
                            + BEFF[c * V_V + v];
            if (ok) outn[(size_t)c * TV + col] = val;
        }
    }
}

extern "C" void kernel_launch(void* const* d_in, const int* in_sizes, int n_in,
                              void* d_out, int out_size, void* d_ws, size_t ws_size,
                              hipStream_t stream) {
    const float* x     = (const float*)d_in[0];   // (64,64,300,25)
    const float* A     = (const float*)d_in[1];   // (3,25,25)
    const float* W     = (const float*)d_in[2];   // (384,64)
    const float* bias  = (const float*)d_in[3];   // (384,)
    const int*   index = (const int*)d_in[4];     // (25,)
    float* out = (float*)d_out;                   // (64,128,300,25)

    float* S    = (float*)d_ws;                   // 75 floats (pad to 128)
    float* BEFF = S + 128;                        // 128*25 floats

    sgc_prep<<<1, 256, 0, stream>>>(A, bias, index, S, BEFF);

    dim3 grid(118, N_N, 1);                       // 118*4 >= 469 column tiles
    sgc_main<<<grid, 256, 0, stream>>>(x, W, S, BEFF, out);
}
